// HiResPrecipNet_wce_new_88330297410063
// MI455X (gfx1250) — matrix operand order repaired and checked by reading, not compiled, not run on hardware
//
#include <hip/hip_runtime.h>
#include <hip/hip_bf16.h>
#include <math.h>

#define NLOW   20000
#define NHIGH  60000
#define ELOW   320000
#define EL2H   540000
#define EHIGH  480000

typedef __attribute__((ext_vector_type(16))) __bf16 v16bf;
typedef __attribute__((ext_vector_type(8)))  float  v8f;

union BFrag { v16bf v; unsigned short u[16]; };

__device__ __forceinline__ unsigned short f2bf(float f) {
    // round-to-nearest-even f32 -> bf16
    unsigned u = __float_as_uint(f);
    u += 0x7FFFu + ((u >> 16) & 1u);
    return (unsigned short)(u >> 16);
}

// ---------------------------------------------------------------------------
// GEMM: C[M,N] = A[M,K] @ W[K,N], f32 in (bf16-rounded RNE), f32 out.
// One wave computes a 16x32 output tile: one A fragment, two B fragments,
// two v_wmma_f32_16x16x32_bf16 per K-step. Block = 128 threads = 4 waves in M.
// Fast path (wave-uniform full tile): unguarded loads, zero exec juggling.
// Ragged path: clamped addresses + cndmask zero-select (branchless).
// ---------------------------------------------------------------------------
__global__ __launch_bounds__(128)
void gemm_wmma_bf16(const float* __restrict__ A, const float* __restrict__ W,
                    float* __restrict__ C, int M, int K, int N) {
    const int lane = threadIdx.x & 31;
    const int wave = threadIdx.x >> 5;
    const int m0   = (blockIdx.x * 4 + wave) * 16;
    const int n0   = blockIdx.y * 32;
    const int half = lane >> 4;     // 0: lanes 0-15, 1: lanes 16-31
    const int mr   = lane & 15;
    const int m    = m0 + mr;       // A row owned by this lane
    const int n    = n0 + mr;       // B col (first 16-wide tile)
    const int n2   = n + 16;        // B col (second tile)

    const bool full_mn = (m0 + 16 <= M) && (n0 + 32 <= N);  // wave-uniform
    const int  mc  = m  < M ? m  : M - 1;
    const int  nc  = n  < N ? n  : N - 1;
    const int  n2c = n2 < N ? n2 : N - 1;

    v8f acc0 = {}, acc1 = {};
    const int kt = (K + 31) >> 5;
    for (int t = 0; t < kt; ++t) {
        const int k0 = t << 5;
        BFrag a, b0, b1;
        if (full_mn && (k0 + 32 <= K)) {            // wave-uniform branch
            const float* pa = A + (size_t)m * K + k0 + half * 16;
            if (k0 + 64 <= K) __builtin_prefetch(pa + 32, 0, 0); // global_prefetch_b8
#pragma unroll
            for (int j = 0; j < 8; ++j) {
                a.u[j]     = f2bf(pa[j]);           // K = k0 + half*16 + j
                a.u[j + 8] = f2bf(pa[8 + j]);       // K = k0 + half*16 + 8 + j
            }
            const float* pb = W + (size_t)(k0 + half * 16) * N + n;
#pragma unroll
            for (int e = 0; e < 16; ++e) {          // K = k0 + half*16 + e
                b0.u[e] = f2bf(pb[(size_t)e * N]);
                b1.u[e] = f2bf(pb[(size_t)e * N + 16]);
            }
        } else {                                    // branchless ragged path
#pragma unroll
            for (int e = 0; e < 16; ++e) {
                int ka  = k0 + half * 16 + e;
                int kac = ka < K ? ka : K - 1;
                float va = A[(size_t)mc * K + kac];
                a.u[e] = (m < M && ka < K) ? f2bf(va) : (unsigned short)0;
                int kb  = k0 + half * 16 + e;
                int kbc = kb < K ? kb : K - 1;
                float v0 = W[(size_t)kbc * N + nc];
                float v1 = W[(size_t)kbc * N + n2c];
                b0.u[e] = (kb < K && n  < N) ? f2bf(v0) : (unsigned short)0;
                b1.u[e] = (kb < K && n2 < N) ? f2bf(v1) : (unsigned short)0;
            }
        }
        acc0 = __builtin_amdgcn_wmma_f32_16x16x32_bf16(false, a.v, false, b0.v,
                                                       (short)0, acc0, false, false);
        acc1 = __builtin_amdgcn_wmma_f32_16x16x32_bf16(false, a.v, false, b1.v,
                                                       (short)0, acc1, false, false);
    }
    // C/D layout: VGPR r -> row m0 + half*8 + r, col n0 + (lane&15)
#pragma unroll
    for (int r = 0; r < 8; ++r) {
        int row = m0 + half * 8 + r;
        if (row < M && n  < N) C[(size_t)row * N + n]  = acc0[r];
        if (row < M && n2 < N) C[(size_t)row * N + n2] = acc1[r];
    }
}

// ---------------------------------------------------------------------------
// Utility / GAT edge kernels
// ---------------------------------------------------------------------------
__global__ void fill_kernel(float* __restrict__ p, float v, size_t n) {
    size_t i = (size_t)blockIdx.x * blockDim.x + threadIdx.x;
    if (i < n) p[i] = v;
}

__device__ __forceinline__ void atomicMaxF(float* addr, float val) {
    // bit-order trick: signed-int max for >=0, unsigned min for <0 (init is -inf)
    if (val >= 0.f) atomicMax((int*)addr, __float_as_int(val));
    else            atomicMin((unsigned int*)addr, __float_as_uint(val));
}

// pass 1: one wave per (edge, head): logit = att . leakyrelu(xl[src]+xr[dst]); seg-max
__global__ __launch_bounds__(256)
void gat_edge_logits(const float* __restrict__ xl, const float* __restrict__ xr,
                     const int* __restrict__ src, const int* __restrict__ dst,
                     const float* __restrict__ att, float* __restrict__ logits,
                     float* __restrict__ mbuf, int E, int H, int C) {
    long long gw = ((long long)blockIdx.x * blockDim.x + threadIdx.x) >> 5;
    int lane = threadIdx.x & 31;
    if (gw >= (long long)E * H) return;
    int e = (int)(gw / H), h = (int)(gw - (long long)e * H);
    int s = src[e], d = dst[e];
    int HC = H * C;
    const float* pl = xl + (size_t)s * HC + h * C;
    const float* pr = xr + (size_t)d * HC + h * C;
    const float* pa = att + h * C;
    float accv = 0.f;
    if (C == 128) {           // lane owns 4 contiguous channels -> b128 loads
        float4 vl = ((const float4*)pl)[lane];
        float4 vr = ((const float4*)pr)[lane];
        float4 va = ((const float4*)pa)[lane];
        float z0 = vl.x + vr.x, z1 = vl.y + vr.y, z2 = vl.z + vr.z, z3 = vl.w + vr.w;
        z0 = z0 > 0.f ? z0 : 0.2f * z0;  z1 = z1 > 0.f ? z1 : 0.2f * z1;
        z2 = z2 > 0.f ? z2 : 0.2f * z2;  z3 = z3 > 0.f ? z3 : 0.2f * z3;
        accv = z0 * va.x + z1 * va.y + z2 * va.z + z3 * va.w;
    } else {
        for (int c = lane; c < C; c += 32) {
            float z = pl[c] + pr[c];
            z = z > 0.f ? z : 0.2f * z;
            accv += z * pa[c];
        }
    }
#pragma unroll
    for (int off = 16; off; off >>= 1) accv += __shfl_xor(accv, off, 32);
    if (lane == 0) {
        logits[gw] = accv;
        atomicMaxF(mbuf + (size_t)d * H + h, accv);
    }
}

// pass 2: one thread per (edge, head): e = exp(logit - m[dst]); seg-sum; deg count
__global__ void gat_edge_soft(const int* __restrict__ dst, float* __restrict__ el,
                              const float* __restrict__ m, float* __restrict__ s,
                              float* __restrict__ deg, int E, int H) {
    long long i = (long long)blockIdx.x * blockDim.x + threadIdx.x;
    if (i >= (long long)E * H) return;
    int e = (int)(i / H), h = (int)(i - (long long)e * H);
    int d = dst[e];
    float ex = __expf(el[i] - m[(size_t)d * H + h]);
    el[i] = ex;
    atomicAdd(&s[(size_t)d * H + h], ex);
    if (h == 0) atomicAdd(&deg[d], 1.0f);
}

// pass 3: one wave per (edge, head): out[dst] += alpha * xl[src]
__global__ __launch_bounds__(256)
void gat_edge_scatter(const float* __restrict__ xl, const int* __restrict__ src,
                      const int* __restrict__ dst, const float* __restrict__ el,
                      const float* __restrict__ s, float* __restrict__ out,
                      int E, int H, int C) {
    long long gw = ((long long)blockIdx.x * blockDim.x + threadIdx.x) >> 5;
    int lane = threadIdx.x & 31;
    if (gw >= (long long)E * H) return;
    int e = (int)(gw / H), h = (int)(gw - (long long)e * H);
    int sn = src[e], d = dst[e];
    int HC = H * C;
    float alpha = el[gw] / s[(size_t)d * H + h];
    const float* pl = xl + (size_t)sn * HC + h * C;
    float* po = out + (size_t)d * HC + h * C;
    if (C == 128) {
        float4 v = ((const float4*)pl)[lane];
        float* q = po + lane * 4;
        atomicAdd(q + 0, v.x * alpha);
        atomicAdd(q + 1, v.y * alpha);
        atomicAdd(q + 2, v.z * alpha);
        atomicAdd(q + 3, v.w * alpha);
    } else {
        for (int c = lane; c < C; c += 32) atomicAdd(&po[c], pl[c] * alpha);
    }
}

// finalize: out = out / max(deg,1) + bias
__global__ void gat_finalize(float* __restrict__ out, const float* __restrict__ deg,
                             const float* __restrict__ bias, int n_dst, int HC) {
    long long i = (long long)blockIdx.x * blockDim.x + threadIdx.x;
    if (i >= (long long)n_dst * HC) return;
    int nidx = (int)(i / HC);
    float d = deg[nidx]; d = d > 1.f ? d : 1.f;
    float v = out[i] / d;
    if (bias) v += bias[i % HC];
    out[i] = v;
}

// BatchNorm: per-channel mean/var (biased, matches x.var(axis=0)), then normalize
__global__ __launch_bounds__(256)
void bn_stats(const float* __restrict__ x, float* __restrict__ stats, int Nrows, int F) {
    int ch = blockIdx.x;
    float s = 0.f, s2 = 0.f;
    for (int r = threadIdx.x; r < Nrows; r += blockDim.x) {
        float v = x[(size_t)r * F + ch];
        s += v; s2 += v * v;
    }
    __shared__ float sh[256], sh2[256];
    sh[threadIdx.x] = s; sh2[threadIdx.x] = s2; __syncthreads();
    for (int o = 128; o; o >>= 1) {
        if (threadIdx.x < o) { sh[threadIdx.x] += sh[threadIdx.x + o]; sh2[threadIdx.x] += sh2[threadIdx.x + o]; }
        __syncthreads();
    }
    if (threadIdx.x == 0) {
        float mu = sh[0] / Nrows;
        stats[ch] = mu;
        stats[F + ch] = sh2[0] / Nrows - mu * mu;
    }
}

__global__ void bn_apply(float* __restrict__ x, const float* __restrict__ stats,
                         const float* __restrict__ g, const float* __restrict__ b,
                         int Nrows, int F) {
    long long i = (long long)blockIdx.x * blockDim.x + threadIdx.x;
    if (i >= (long long)Nrows * F) return;
    int ch = (int)(i % F);
    x[i] = (x[i] - stats[ch]) * rsqrtf(stats[F + ch] + 1e-5f) * g[ch] + b[ch];
}

// concat [z_std | eh(128)] -> 129 cols
__global__ void concat_kernel(const float* __restrict__ z, const float* __restrict__ eh,
                              float* __restrict__ out, int Nrows) {
    long long i = (long long)blockIdx.x * blockDim.x + threadIdx.x;
    if (i >= (long long)Nrows * 129) return;
    int r = (int)(i / 129), c = (int)(i - (long long)r * 129);
    out[i] = (c == 0) ? z[r] : eh[(size_t)r * 128 + (c - 1)];
}

// ---------------------------------------------------------------------------
// Host orchestration
// ---------------------------------------------------------------------------
extern "C" void kernel_launch(void* const* d_in, const int* in_sizes, int n_in,
                              void* d_out, int out_size, void* d_ws, size_t ws_size,
                              hipStream_t stream) {
    const float* x_low  = (const float*)d_in[0];
    const float* x_high = (const float*)d_in[1];
    const float* z_std  = (const float*)d_in[2];
    const int*   e_low  = (const int*)d_in[3];
    const int*   e_l2h  = (const int*)d_in[4];
    const int*   e_high = (const int*)d_in[5];
    const float *l1_Wl = (const float*)d_in[6],  *l1_Wr = (const float*)d_in[7];
    const float *l1_att = (const float*)d_in[8], *l1_b = (const float*)d_in[9];
    const float *bn1_g = (const float*)d_in[10], *bn1_b = (const float*)d_in[11];
    const float *l2_Wl = (const float*)d_in[12], *l2_Wr = (const float*)d_in[13];
    const float *l2_att = (const float*)d_in[14], *l2_b = (const float*)d_in[15];
    const float *bn2_g = (const float*)d_in[16], *bn2_b = (const float*)d_in[17];
    const float *l3_Wl = (const float*)d_in[18], *l3_Wr = (const float*)d_in[19];
    const float *l3_att = (const float*)d_in[20], *l3_b = (const float*)d_in[21];
    const float *lh_Wl = (const float*)d_in[22], *lh_Wr = (const float*)d_in[23];
    const float *lh_att = (const float*)d_in[24];
    const float *h1_Wl = (const float*)d_in[25], *h1_Wr = (const float*)d_in[26];
    const float *h1_att = (const float*)d_in[27], *h1_b = (const float*)d_in[28];
    const float *bn3_g = (const float*)d_in[29], *bn3_b = (const float*)d_in[30];
    const float *h2_Wl = (const float*)d_in[31], *h2_Wr = (const float*)d_in[32];
    const float *h2_att = (const float*)d_in[33], *h2_b = (const float*)d_in[34];
    const float *bn4_g = (const float*)d_in[35], *bn4_b = (const float*)d_in[36];
    const float *h3_Wl = (const float*)d_in[37], *h3_Wr = (const float*)d_in[38];
    const float *h3_att = (const float*)d_in[39], *h3_b = (const float*)d_in[40];

    // workspace layout (~251 MB): 4 rotating node buffers + softmax scratch
    const size_t BIG = (size_t)NHIGH * 256;
    float* w     = (float*)d_ws;
    float* BA    = w;
    float* BB    = BA + BIG;
    float* BC    = BB + BIG;
    float* BD    = BC + BIG;
    float* ELb   = BD + BIG;        // max E*H = 960000
    float* Mbuf  = ELb + 960000;    // 60000*2
    float* Sbuf  = Mbuf + 120000;
    float* DEG   = Sbuf + 120000;   // 60000
    float* STATS = DEG + 60000;     // 2*256

    auto gemm = [&](const float* A, const float* W, float* Cout, int M, int K, int N) {
        dim3 g((unsigned)((M + 63) / 64), (unsigned)((N + 31) / 32));
        gemm_wmma_bf16<<<g, 128, 0, stream>>>(A, W, Cout, M, K, N);
    };
    auto fill = [&](float* p, float v, size_t n) {
        fill_kernel<<<(unsigned)((n + 255) / 256), 256, 0, stream>>>(p, v, n);
    };
    auto gat = [&](const float* XL, const float* XR, const int* eidx, int E,
                   int n_dst, int H, int C, const float* att, const float* bias, float* OUT) {
        const int* src = eidx;
        const int* dst = eidx + E;
        int HC = H * C;
        fill(Mbuf, -__builtin_inff(), (size_t)n_dst * H);
        fill(Sbuf, 0.f, (size_t)n_dst * H);
        fill(DEG, 0.f, (size_t)n_dst);
        fill(OUT, 0.f, (size_t)n_dst * HC);
        long long waves = (long long)E * H;
        long long t1 = waves * 32;
        gat_edge_logits<<<(unsigned)((t1 + 255) / 256), 256, 0, stream>>>(
            XL, XR, src, dst, att, ELb, Mbuf, E, H, C);
        gat_edge_soft<<<(unsigned)((waves + 255) / 256), 256, 0, stream>>>(
            dst, ELb, Mbuf, Sbuf, DEG, E, H);
        gat_edge_scatter<<<(unsigned)((t1 + 255) / 256), 256, 0, stream>>>(
            XL, src, dst, ELb, Sbuf, OUT, E, H, C);
        gat_finalize<<<(unsigned)(((long long)n_dst * HC + 255) / 256), 256, 0, stream>>>(
            OUT, DEG, bias, n_dst, HC);
    };
    auto bn = [&](float* X, const float* g, const float* b, int Nrows, int F) {
        bn_stats<<<(unsigned)F, 256, 0, stream>>>(X, STATS, Nrows, F);
        bn_apply<<<(unsigned)(((long long)Nrows * F + 255) / 256), 256, 0, stream>>>(
            X, STATS, g, b, Nrows, F);
    };

    // low_net layer 1: GATv2(125->128, H=2) + BN(256)
    gemm(x_low, l1_Wl, BA, NLOW, 125, 256);
    gemm(x_low, l1_Wr, BB, NLOW, 125, 256);
    gat(BA, BB, e_low, ELOW, NLOW, 2, 128, l1_att, l1_b, BC);
    bn(BC, bn1_g, bn1_b, NLOW, 256);
    // layer 2: GATv2(256->128) + BN(128)
    gemm(BC, l2_Wl, BA, NLOW, 256, 128);
    gemm(BC, l2_Wr, BB, NLOW, 256, 128);
    gat(BA, BB, e_low, ELOW, NLOW, 1, 128, l2_att, l2_b, BD);
    bn(BD, bn2_g, bn2_b, NLOW, 128);
    // layer 3: GATv2(128->128)
    gemm(BD, l3_Wl, BA, NLOW, 128, 128);
    gemm(BD, l3_Wr, BB, NLOW, 128, 128);
    gat(BA, BB, e_low, ELOW, NLOW, 1, 128, l3_att, l3_b, BC);   // BC = h_low [20000,128]
    // bipartite low2high: xl = h_low @ lh_Wl, xr = x_high @ lh_Wr, no bias
    gemm(BC, lh_Wl, BA, NLOW, 128, 128);
    gemm(x_high, lh_Wr, BB, NHIGH, 1, 128);
    gat(BA, BB, e_l2h, EL2H, NHIGH, 1, 128, lh_att, nullptr, BD); // BD = eh [60000,128]
    concat_kernel<<<(unsigned)(((long long)NHIGH * 129 + 255) / 256), 256, 0, stream>>>(
        z_std, BD, BC, NHIGH);                                    // BC = [60000,129]
    // high_net layer 1: GATv2(129->128, H=2) + BN(256)
    gemm(BC, h1_Wl, BA, NHIGH, 129, 256);
    gemm(BC, h1_Wr, BB, NHIGH, 129, 256);
    gat(BA, BB, e_high, EHIGH, NHIGH, 2, 128, h1_att, h1_b, BD);
    bn(BD, bn3_g, bn3_b, NHIGH, 256);
    // layer 2: GATv2(256->128) + BN(128)
    gemm(BD, h2_Wl, BA, NHIGH, 256, 128);
    gemm(BD, h2_Wr, BB, NHIGH, 256, 128);
    gat(BA, BB, e_high, EHIGH, NHIGH, 1, 128, h2_att, h2_b, BC);
    bn(BC, bn4_g, bn4_b, NHIGH, 128);
    // layer 3: GATv2(128->2) -> d_out
    gemm(BC, h3_Wl, BA, NHIGH, 128, 2);
    gemm(BC, h3_Wr, BB, NHIGH, 128, 2);
    gat(BA, BB, e_high, EHIGH, NHIGH, 1, 2, h3_att, h3_b, (float*)d_out);
}
